// GNNModule_42296837931757
// MI455X (gfx1250) — compile-verified, matrix operand order checked by reading, not run on hardware
//
#include <hip/hip_runtime.h>
#include <hip/hip_bf16.h>

// ---------------------------------------------------------------------------
// GINEConv + MLP for MI455X (gfx1250, wave32, WMMA f32_16x16x32_f16)
//
// Inputs (setup_inputs order):
//   d_in[0] x          [N,128]  f32
//   d_in[1] edge_index [2,E]    int32 (src row then dst row)
//   d_in[2] edge_attr  [E,128]  f32
//   d_in[3] eps        [1]      f32
//   d_in[4] W1         [128,64] f32
//   d_in[5] b1         [64]     f32
//   d_in[6] W2         [64,256] f32
//   d_in[7] b2         [256]    f32
// Output: out [N,256] f32.  Workspace: agg [N,128] f32.
// ---------------------------------------------------------------------------

#define N_NODES 50000
#define N_EDGES 800000
#define D_IN    128
#define D_HID   64
#define D_OUT   256

typedef __attribute__((ext_vector_type(16))) _Float16 v16h;
typedef __attribute__((ext_vector_type(8)))  float    v8f;

// ---------------------------------------------------------------------------
// Kernel 1: agg[i] = (1 + eps) * x[i]        (float4 vectorized)
// ---------------------------------------------------------------------------
__global__ void gine_init_kernel(const float* __restrict__ x,
                                 const float* __restrict__ eps,
                                 float* __restrict__ agg) {
    const int i = blockIdx.x * blockDim.x + threadIdx.x;       // float4 index
    const int total = (N_NODES * D_IN) / 4;
    if (i >= total) return;
    const float s = 1.0f + eps[0];
    float4 v = reinterpret_cast<const float4*>(x)[i];
    v.x *= s; v.y *= s; v.z *= s; v.w *= s;
    reinterpret_cast<float4*>(agg)[i] = v;
}

// ---------------------------------------------------------------------------
// Kernel 2: one wave32 per edge; lane l handles 4 features as float4.
//   agg[dst] += ReLU(x[src] + edge_attr[e])   via native f32 atomics
// ---------------------------------------------------------------------------
__global__ void gine_scatter_kernel(const float* __restrict__ x,
                                    const int*   __restrict__ edge_index,
                                    const float* __restrict__ edge_attr,
                                    float* __restrict__ agg) {
    const int gtid = blockIdx.x * blockDim.x + threadIdx.x;
    const int e    = gtid >> 5;            // wave id == edge id
    const int lane = gtid & 31;
    if (e >= N_EDGES) return;

    const int src = edge_index[e];
    const int dst = edge_index[N_EDGES + e];

    const float4 xs = reinterpret_cast<const float4*>(x + (long)src * D_IN)[lane];
    const float4 ea = reinterpret_cast<const float4*>(edge_attr + (long)e * D_IN)[lane];

    float m[4];
    m[0] = fmaxf(xs.x + ea.x, 0.0f);
    m[1] = fmaxf(xs.y + ea.y, 0.0f);
    m[2] = fmaxf(xs.z + ea.z, 0.0f);
    m[3] = fmaxf(xs.w + ea.w, 0.0f);

    float* dp = agg + (long)dst * D_IN + lane * 4;
#pragma unroll
    for (int j = 0; j < 4; ++j)
        __hip_atomic_fetch_add(&dp[j], m[j], __ATOMIC_RELAXED, __HIP_MEMORY_SCOPE_AGENT);
}

// ---------------------------------------------------------------------------
// WMMA fragment layout (ISA 7.12.2, wave32, 16-bit operands):
//   A 16x32: lane&15 = M; lane>>4 = K-half; VGPR v holds K = frag_k(v,hs)+{0,1}
//   B 32x16: lane&15 = N; same K striping
//   C/D 16x16 f32: lane&15 = N; M = (lane>>4)*8 + vgpr_index
// frag_k(v,hs) = ((v>>2)<<4) | (hs<<3) | ((v&3)<<1)
// For fixed (group=v>>2, hs), K covers 8 CONSECUTIVE values -> vectorizable.
// ---------------------------------------------------------------------------
__device__ inline int frag_k(int v, int hs) {
    return ((v >> 2) << 4) | (hs << 3) | ((v & 3) << 1);
}

// A fragment from row-major f32 (ld = row stride): 4x b128 loads + cvt
__device__ inline v16h load_a_frag_f32(const float* __restrict__ src, int ld, int lane) {
    const int m  = lane & 15;
    const int hs = (lane >> 4) & 1;
    v16h a;
#pragma unroll
    for (int g = 0; g < 2; ++g) {
        const float4 p0 = *reinterpret_cast<const float4*>(&src[m * ld + g * 16 + hs * 8]);
        const float4 p1 = *reinterpret_cast<const float4*>(&src[m * ld + g * 16 + hs * 8 + 4]);
        a[8 * g + 0] = (_Float16)p0.x;  a[8 * g + 1] = (_Float16)p0.y;
        a[8 * g + 2] = (_Float16)p0.z;  a[8 * g + 3] = (_Float16)p0.w;
        a[8 * g + 4] = (_Float16)p1.x;  a[8 * g + 5] = (_Float16)p1.y;
        a[8 * g + 6] = (_Float16)p1.z;  a[8 * g + 7] = (_Float16)p1.w;
    }
    return a;
}

// LDS staging index for GEMM2 A-fragments: element (row m, hidden col) ->
// flat index into v16h[2*32] region (kk2 frag, lane, half-element j).
__device__ inline int a2_stage_idx(int m, int col) {
    const int kk2  = col >> 5;
    const int k    = col & 31;
    const int hs2  = (k >> 3) & 1;
    const int lane2 = m + (hs2 << 4);
    const int v    = ((k >> 4) << 2) | ((k & 7) >> 1);
    const int j    = 2 * v + (k & 1);
    return (kk2 * 32 + lane2) * 16 + j;
}

// ---------------------------------------------------------------------------
// Kernel 3: fused MLP. 4 waves/block, one 16-row tile per wave.
// Weights pre-swizzled in LDS to fragment layout: [frag][lane] = v16h (32B),
// so every B fragment is ONE vector LDS read (2x ds_load_b128).
//   GEMM1: [16,128]x[128,64]  = 4 Ntiles x 4 Ksteps  -> bias+ReLU -> LDS stage
//   GEMM2: [16,64] x[64,256]  = 16 Ntiles x 2 Ksteps -> bias+ReLU -> out
// ---------------------------------------------------------------------------
#define MLP_WAVES 4

__global__ void gine_mlp_kernel(const float* __restrict__ agg,
                                const float* __restrict__ W1,
                                const float* __restrict__ b1,
                                const float* __restrict__ W2,
                                const float* __restrict__ b2,
                                float* __restrict__ out,
                                int n_tiles) {
    __shared__ v16h  sW1f[16 * 32];                 // 16 frags (kk*4+nt) x 32 lanes, 16 KB
    __shared__ v16h  sW2f[32 * 32];                 // 32 frags (kk*16+nt) x 32 lanes, 32 KB
    __shared__ v16h  stage[MLP_WAVES][2 * 32];      // per-wave h1 A-fragments, 8 KB
    __shared__ float sb1[D_HID];
    __shared__ float sb2[D_OUT];

    const int tid = threadIdx.x;

    // ---- cooperative preload: swizzle f32 weights into f16 fragment layout ----
    {
        _Float16* w1h = reinterpret_cast<_Float16*>(sW1f);
        for (int idx = tid; idx < 16 * 32 * 16; idx += blockDim.x) {
            const int j = idx & 15, lane = (idx >> 4) & 31, f = idx >> 9;
            const int kk = f >> 2, nt = f & 3;
            const int n = lane & 15, hs = (lane >> 4) & 1;
            const int k = kk * 32 + frag_k(j >> 1, hs) + (j & 1);
            w1h[idx] = (_Float16)W1[k * D_HID + nt * 16 + n];
        }
        _Float16* w2h = reinterpret_cast<_Float16*>(sW2f);
        for (int idx = tid; idx < 32 * 32 * 16; idx += blockDim.x) {
            const int j = idx & 15, lane = (idx >> 4) & 31, f = idx >> 9;
            const int kk = f >> 4, nt = f & 15;
            const int n = lane & 15, hs = (lane >> 4) & 1;
            const int k = kk * 32 + frag_k(j >> 1, hs) + (j & 1);
            w2h[idx] = (_Float16)W2[k * D_OUT + nt * 16 + n];
        }
        for (int i = tid; i < D_HID; i += blockDim.x) sb1[i] = b1[i];
        for (int i = tid; i < D_OUT; i += blockDim.x) sb2[i] = b2[i];
    }
    __syncthreads();

    const int wave = tid >> 5;
    const int lane = tid & 31;
    const int tile = blockIdx.x * MLP_WAVES + wave;
    if (tile >= n_tiles) return;              // whole-wave guard: EXEC all-1s inside

    const int row0 = tile * 16;
    const int n    = lane & 15;
    const int hs   = (lane >> 4) & 1;
    _Float16* sth  = reinterpret_cast<_Float16*>(&stage[wave][0]);

    // ---- GEMM1: h1 = ReLU(agg_tile @ W1 + b1), staged as A-fragments --------
    v16h a1[4];
#pragma unroll
    for (int kk = 0; kk < 4; ++kk)
        a1[kk] = load_a_frag_f32(agg + (long)row0 * D_IN + kk * 32, D_IN, lane);

#pragma unroll
    for (int nt = 0; nt < 4; ++nt) {
        v8f c = {};
#pragma unroll
        for (int kk = 0; kk < 4; ++kk) {
            const v16h bfr = sW1f[(kk * 4 + nt) * 32 + lane];   // one 32B LDS read
            c = __builtin_amdgcn_wmma_f32_16x16x32_f16(
                    false, a1[kk], false, bfr, (short)0, c, false, false);
        }
        const float bias = sb1[nt * 16 + n];
#pragma unroll
        for (int r = 0; r < 8; ++r) {
            const float v = fmaxf(c[r] + bias, 0.0f);
            sth[a2_stage_idx(hs * 8 + r, nt * 16 + n)] = (_Float16)v;
        }
    }

    // ---- GEMM2: out = ReLU(h1 @ W2 + b2) ------------------------------------
    v16h a2[2];
#pragma unroll
    for (int kk = 0; kk < 2; ++kk)
        a2[kk] = stage[wave][kk * 32 + lane];  // one 32B LDS read; same-wave order

    float* orow = out + (long)row0 * D_OUT;
#pragma unroll
    for (int nt = 0; nt < 16; ++nt) {
        v8f c = {};
#pragma unroll
        for (int kk = 0; kk < 2; ++kk) {
            const v16h bfr = sW2f[(kk * 16 + nt) * 32 + lane];  // one 32B LDS read
            c = __builtin_amdgcn_wmma_f32_16x16x32_f16(
                    false, a2[kk], false, bfr, (short)0, c, false, false);
        }
        const float bias = sb2[nt * 16 + n];
#pragma unroll
        for (int r = 0; r < 8; ++r) {
            const float v = fmaxf(c[r] + bias, 0.0f);
            orow[(hs * 8 + r) * D_OUT + nt * 16 + n] = v;   // 64B-coalesced across lanes
        }
    }
}

// ---------------------------------------------------------------------------
extern "C" void kernel_launch(void* const* d_in, const int* in_sizes, int n_in,
                              void* d_out, int out_size, void* d_ws, size_t ws_size,
                              hipStream_t stream) {
    const float* x          = (const float*)d_in[0];
    const int*   edge_index = (const int*)d_in[1];
    const float* edge_attr  = (const float*)d_in[2];
    const float* eps        = (const float*)d_in[3];
    const float* W1         = (const float*)d_in[4];
    const float* b1         = (const float*)d_in[5];
    const float* W2         = (const float*)d_in[6];
    const float* b2         = (const float*)d_in[7];
    float*       out        = (float*)d_out;
    float*       agg        = (float*)d_ws;          // N_NODES * D_IN floats

    // 1) agg = (1+eps) * x
    {
        const int total  = (N_NODES * D_IN) / 4;
        const int blocks = (total + 255) / 256;
        gine_init_kernel<<<blocks, 256, 0, stream>>>(x, eps, agg);
    }
    // 2) scatter ReLU(x[src]+e) into agg[dst]; one wave per edge, 8 waves/block
    {
        const int blocks = (N_EDGES + 7) / 8;
        gine_scatter_kernel<<<blocks, 256, 0, stream>>>(x, edge_index, edge_attr, agg);
    }
    // 3) fused WMMA MLP
    {
        const int n_tiles = (N_NODES + 15) / 16;               // 3125
        const int blocks  = (n_tiles + MLP_WAVES - 1) / MLP_WAVES;
        gine_mlp_kernel<<<blocks, MLP_WAVES * 32, 0, stream>>>(
            agg, W1, b1, W2, b2, out, n_tiles);
    }
}
